// IntegratorImplicitShortHorizon_57629871178189
// MI455X (gfx1250) — compile-verified, matrix operand order checked by reading, not run on hardware
//
#include <hip/hip_runtime.h>
#include <hip/hip_bf16.h>

// ---------------------------------------------------------------------------
// Kvaerno5 ESDIRK step, batch=256, NX=128, NH=256, NU=16.
// Newton solve replaced by 4-term Neumann series (||h*gamma*J|| ~ 0.011, so
// truncation error ~1e-8 rel): dk = r + Xr + X^2 r + X^3 r, X = h*g*J,
// J*v = -LEAK*v + W2 (d .* (W1 v))  -> pure batched matvecs -> WMMA GEMMs.
// Weights staged once per block into LDS via global_load_async_to_lds_b128
// (ASYNCcnt) so the ~100 sequential GEMM pairs hit LDS instead of L2.
// ---------------------------------------------------------------------------

typedef __attribute__((ext_vector_type(16))) _Float16 v16h;
typedef __attribute__((ext_vector_type(8)))  _Float16 v8h;
typedef __attribute__((ext_vector_type(8)))  float    v8f;
typedef __attribute__((ext_vector_type(4)))  int      v4i;

typedef __attribute__((address_space(1))) v4i* gptr_v4i;  // global int4*
typedef __attribute__((address_space(3))) v4i* lptr_v4i;  // LDS int4*

#define NXD 128
#define NHD 256
#define NUD 16
#define BT  16      // batch rows per block
#define NT  256     // threads per block (8 wave32)
#define DT_F   0.02f
#define GAM_F  0.26f
#define LEAK_F 0.1f
#define HG     (DT_F * GAM_F)

#define NFRAG_HALVES 65536   // 32768 W1^T frags + 32768 W2^T frags (f16)

__constant__ float c_alow[6][6] = {
  {0.26f, 0.f, 0.f, 0.f, 0.f, 0.f},
  {0.13f, 0.84033320996790809f, 0.f, 0.f, 0.f, 0.f},
  {0.22371961478320505f, 0.47675532319799699f, -0.06470895363112615f, 0.f, 0.f, 0.f},
  {0.16648564323248321f, 0.1045001884159172f, 0.03631482272098715f, -0.13090704451073998f, 0.f, 0.f},
  {0.13855640231268224f, 0.0f, -0.04245337201752043f, 0.02446657898003141f, 0.61943039072480676f, 0.f},
  {0.13659751177640291f, 0.0f, -0.05496908796538376f, -0.04118626728321046f, 0.62993304899016403f, 0.06962479448202728f}
};
__constant__ float c_bsol[7] = {
  0.13659751177640291f, 0.0f, -0.05496908796538376f, -0.04118626728321046f,
  0.62993304899016403f, 0.06962479448202728f, 0.26f
};

// K-index mapping for 16-bit A/B WMMA fragments (16x32 / 32x16), wave32.
static __device__ __forceinline__ int kmap(int e, int lane) {
  return ((lane & 16) ? 8 : 0) + ((e >> 3) << 4) + (e & 7);
}

// Load A fragment (16xK tile, row = batch index) from row-major f16 LDS staging.
static __device__ __forceinline__ v16h load_a(const _Float16* st, int ld, int kb, int lane) {
  const int row   = lane & 15;
  const int kbase = kb * 32 + ((lane & 16) ? 8 : 0);
  const _Float16* p = st + row * ld + kbase;
  v8h lo = *(const v8h*)(p);        // K = kbase .. kbase+7
  v8h hi = *(const v8h*)(p + 16);   // K = kbase+16 .. kbase+23
  v16h a;
#pragma unroll
  for (int e = 0; e < 8; ++e) { a[e] = lo[e]; a[e + 8] = hi[e]; }
  return a;
}

// Load B fragment from pre-swizzled LDS (contiguous 32B per lane).
static __device__ __forceinline__ v16h load_b(const _Float16* frags, int fidx, int lane) {
  return *(const v16h*)(frags + (size_t)(fidx * 32 + lane) * 16);
}

// GEMM1: out[16][256] = A(16x128 from sa) * W1^T  (K=128, 16 N-tiles, 2/wave)
static __device__ __forceinline__ void gemm1(const _Float16* sa, const _Float16* w1f,
                                             float* out, int tid) {
  const int lane = tid & 31, w = tid >> 5;
#pragma unroll
  for (int t = 0; t < 2; ++t) {
    const int nb = w * 2 + t;
    v8f acc = {};
#pragma unroll
    for (int kb = 0; kb < 4; ++kb) {
      v16h a = load_a(sa, NXD, kb, lane);
      v16h b = load_b(w1f, nb * 4 + kb, lane);
      acc = __builtin_amdgcn_wmma_f32_16x16x32_f16(false, a, false, b,
                                                   (short)0, acc, false, false);
    }
    const int row0 = (lane & 16) ? 8 : 0;
    const int col  = nb * 16 + (lane & 15);
#pragma unroll
    for (int r = 0; r < 8; ++r) out[(row0 + r) * NHD + col] = acc[r];
  }
}

// GEMM2: out[16][128] = A(16x256 from sa) * W2^T  (K=256, 8 N-tiles, 1/wave)
static __device__ __forceinline__ void gemm2(const _Float16* sa, const _Float16* w2f,
                                             float* out, int tid) {
  const int lane = tid & 31, nb = tid >> 5;
  v8f acc = {};
#pragma unroll
  for (int kb = 0; kb < 8; ++kb) {
    v16h a = load_a(sa, NHD, kb, lane);
    v16h b = load_b(w2f, nb * 8 + kb, lane);
    acc = __builtin_amdgcn_wmma_f32_16x16x32_f16(false, a, false, b,
                                                 (short)0, acc, false, false);
  }
  const int row0 = (lane & 16) ? 8 : 0;
  const int col  = nb * 16 + (lane & 15);
#pragma unroll
  for (int r = 0; r < 8; ++r) out[(row0 + r) * NXD + col] = acc[r];
}

// ---------------------------------------------------------------------------
// Prep: convert W1/W2 to f16 and swizzle into WMMA B-fragment order in d_ws.
//   ws[0      .. 32767] : W1^T frags  (K=NX=128 -> 4 kb, N=NH=256 -> 16 nb)
//   ws[32768 .. 65535]  : W2^T frags  (K=NH=256 -> 8 kb, N=NX=128 ->  8 nb)
// ---------------------------------------------------------------------------
__global__ void kv5_prep_kernel(const float* __restrict__ W1,
                                const float* __restrict__ W2,
                                _Float16* __restrict__ ws) {
  int idx = blockIdx.x * blockDim.x + threadIdx.x;
  if (idx < 32768) {
    int f = idx >> 9, rem = idx & 511, lane = rem >> 4, e = rem & 15;
    int kb = f & 3, nb = f >> 2;
    int x = kb * 32 + kmap(e, lane);
    int h = nb * 16 + (lane & 15);
    ws[idx] = (_Float16)W1[h * NXD + x];          // B[k=x][n=h] = W1[h][x]
  } else if (idx < NFRAG_HALVES) {
    int j = idx - 32768;
    int f = j >> 9, rem = j & 511, lane = rem >> 4, e = rem & 15;
    int kb = f & 7, nb = f >> 3;
    int hh = kb * 32 + kmap(e, lane);
    int xx = nb * 16 + (lane & 15);
    ws[idx] = (_Float16)W2[xx * NHD + hh];        // B[k=h][n=x] = W2[x][h]
  }
}

// ---------------------------------------------------------------------------
// Main kernel: one block integrates 16 batch rows through the full step.
// ---------------------------------------------------------------------------
__global__ __launch_bounds__(NT) void kv5_step_kernel(
    const float* __restrict__ xin, const float* __restrict__ uin,
    const float* __restrict__ b1,  const float* __restrict__ Bu,
    const float* __restrict__ b2,
    const _Float16* __restrict__ wfr,   // all 65536 fragment halves
    float* __restrict__ out) {
  __shared__ float s_ks[7][BT * NXD];                 // 56 KB stage slopes
  __shared__ float s_y0[BT * NXD];                    // 8 KB
  __shared__ float s_pred[BT * NXD];                  // 8 KB
  __shared__ float s_v[BT * NXD];                     // 8 KB Neumann term
  __shared__ float s_z[BT * NHD];                     // 16 KB GEMM1 out
  __shared__ _Float16 s_dact[BT * NHD];               // 8 KB tanh'
  __shared__ float s_f[BT * NXD];                     // 8 KB GEMM2 out
  __shared__ float s_c1[BT * NHD];                    // 16 KB b1 + Bu*clip(u)
  __shared__ __align__(16) _Float16 s_a16[BT * NXD];  // 4 KB A staging (NX)
  __shared__ __align__(16) _Float16 s_h16[BT * NHD];  // 8 KB A staging (NH)
  __shared__ __align__(32) _Float16 s_wf[NFRAG_HALVES]; // 128 KB weight frags

  const int tid = threadIdx.x;
  const int r0  = blockIdx.x * BT;

  // --- stage all weight fragments into LDS (reused ~100x per block) --------
#if __has_builtin(__builtin_amdgcn_global_load_async_to_lds_b128)
  {
    gptr_v4i gsrc = (gptr_v4i)wfr;    // addrspace(1) int4*
    lptr_v4i ldst = (lptr_v4i)s_wf;   // addrspace(3) int4*
    for (int i = tid; i < NFRAG_HALVES / 8; i += NT) {
      __builtin_amdgcn_global_load_async_to_lds_b128(gsrc + i, ldst + i, 0, 0);
    }
#if __has_builtin(__builtin_amdgcn_s_wait_asynccnt)
    __builtin_amdgcn_s_wait_asynccnt(0);
#else
    asm volatile("s_wait_asynccnt 0" ::: "memory");
#endif
  }
#else
  for (int i = tid; i < NFRAG_HALVES / 8; i += NT)
    ((uint4*)s_wf)[i] = ((const uint4*)wfr)[i];
#endif
  const _Float16* w1f = s_wf;
  const _Float16* w2f = s_wf + 32768;

  // --- init: y0, c1 = b1 + Bu @ clip(u) (u constant over the step) ---------
  for (int e = tid; e < BT * NXD; e += NT) s_y0[e] = xin[r0 * NXD + e];
  for (int e = tid; e < BT * NHD; e += NT) {
    int i = e >> 8, h = e & 255;
    float acc = b1[h];
    const float* br = Bu + h * NUD;
    const float* ur = uin + (r0 + i) * NUD;
#pragma unroll
    for (int j = 0; j < NUD; ++j) {
      float uc = fminf(1.0f, fmaxf(-1.0f, ur[j]));
      acc += br[j] * uc;
    }
    s_c1[e] = acc;
  }
  for (int e = tid; e < BT * NXD; e += NT) s_a16[e] = (_Float16)s_y0[e];
  __syncthreads();

  // --- ks[0] = f(y0) -------------------------------------------------------
  gemm1(s_a16, w1f, s_z, tid);
  __syncthreads();
  for (int e = tid; e < BT * NHD; e += NT) {
    float th = tanhf(s_z[e] + s_c1[e]);
    s_dact[e] = (_Float16)(1.0f - th * th);
    s_h16[e]  = (_Float16)th;
  }
  __syncthreads();
  gemm2(s_h16, w2f, s_f, tid);
  __syncthreads();
  for (int e = tid; e < BT * NXD; e += NT)
    s_ks[0][e] = -LEAK_F * s_y0[e] + s_f[e] + b2[e & (NXD - 1)];
  __syncthreads();

  // --- implicit stages 2..7 ------------------------------------------------
  for (int st = 1; st <= 6; ++st) {
    float* k = s_ks[st];
    for (int e = tid; e < BT * NXD; e += NT) {
      float p = s_y0[e];
      for (int j = 0; j < st; ++j) p += DT_F * c_alow[st - 1][j] * s_ks[j][e];
      s_pred[e] = p;
      k[e] = s_ks[st - 1][e];   // Newton init = previous stage slope
    }
    __syncthreads();

    for (int it = 0; it < 4; ++it) {
      // y_i = pred + h*g*k  (f16 staged for GEMM1)
      for (int e = tid; e < BT * NXD; e += NT)
        s_a16[e] = (_Float16)(s_pred[e] + HG * k[e]);
      __syncthreads();
      gemm1(s_a16, w1f, s_z, tid);
      __syncthreads();
      for (int e = tid; e < BT * NHD; e += NT) {
        float th = tanhf(s_z[e] + s_c1[e]);
        s_dact[e] = (_Float16)(1.0f - th * th);   // J evaluated at y_i
        s_h16[e]  = (_Float16)th;
      }
      __syncthreads();
      gemm2(s_h16, w2f, s_f, tid);
      __syncthreads();
      // r = k - f(y_i);  k -= r  (Neumann term m=0);  stage v=r
      for (int e = tid; e < BT * NXD; e += NT) {
        float yi = s_pred[e] + HG * k[e];
        float fy = -LEAK_F * yi + s_f[e] + b2[e & (NXD - 1)];
        float r  = k[e] - fy;
        k[e]     = fy;
        s_v[e]   = r;
        s_a16[e] = (_Float16)r;
      }
      __syncthreads();
      // Neumann terms m=1..3:  v <- h*g*Jv;  k -= v
      for (int m = 0; m < 3; ++m) {
        gemm1(s_a16, w1f, s_z, tid);             // W1 v
        __syncthreads();
        for (int e = tid; e < BT * NHD; e += NT)
          s_h16[e] = (_Float16)((float)s_dact[e] * s_z[e]);
        __syncthreads();
        gemm2(s_h16, w2f, s_f, tid);             // W2 (d .* W1 v)
        __syncthreads();
        for (int e = tid; e < BT * NXD; e += NT) {
          float nv = HG * (-LEAK_F * s_v[e] + s_f[e]);
          k[e] -= nv;
          s_v[e] = nv;
          s_a16[e] = (_Float16)nv;
        }
        __syncthreads();
      }
    }
  }

  // --- y1 = y0 + h * sum_j b_j ks[j] --------------------------------------
  for (int e = tid; e < BT * NXD; e += NT) {
    float y1 = s_y0[e];
#pragma unroll
    for (int j = 0; j < 7; ++j) y1 += DT_F * c_bsol[j] * s_ks[j][e];
    out[r0 * NXD + e] = y1;
  }
}

// ---------------------------------------------------------------------------
extern "C" void kernel_launch(void* const* d_in, const int* in_sizes, int n_in,
                              void* d_out, int out_size, void* d_ws, size_t ws_size,
                              hipStream_t stream) {
  const float* x  = (const float*)d_in[0];   // [256,128]
  const float* u  = (const float*)d_in[1];   // [256,16]
  const float* W1 = (const float*)d_in[2];   // [256,128]
  const float* b1 = (const float*)d_in[3];   // [256]
  const float* Bu = (const float*)d_in[4];   // [256,16]
  const float* W2 = (const float*)d_in[5];   // [128,256]
  const float* b2 = (const float*)d_in[6];   // [128]
  float* out = (float*)d_out;
  _Float16* ws = (_Float16*)d_ws;            // 128 KB of f16 fragments

  kv5_prep_kernel<<<dim3(256), dim3(256), 0, stream>>>(W1, W2, ws);
  kv5_step_kernel<<<dim3(16), dim3(NT), 0, stream>>>(x, u, b1, Bu, b2, ws, out);
  (void)in_sizes; (void)n_in; (void)out_size; (void)ws_size;
}
